// ClusteringMultiTaskLSTM_87531433493189
// MI455X (gfx1250) — compile-verified, matrix operand order checked by reading, not run on hardware
//
#include <hip/hip_runtime.h>
#include <hip/hip_bf16.h>

// ---------------------------------------------------------------------------
// ClusteringMultiTaskLSTM on MI455X (gfx1250, wave32, WMMA bf16 16x16x32).
//
//  1) pack_weights_kernel : f32 -> bf16 WMMA B-fragment packing (L2 resident)
//  2) pack_bias_kernel    : combine bih+bhh
//  3) encoder_kernel      : 1 wave per (cluster, 16 batch rows); 168 steps of
//                           2-layer LSTM, all GEMMs via v_wmma_f32_16x16x32_bf16,
//                           h recirculated through LDS (D-frag -> A-frag).
//  4) decoder_kernel      : 1 workgroup (512 thr = 16 waves) per feature;
//                           24 steps of LSTM + fc1 + qkv + batch-attention
//                           (256x256 softmax) + out-proj + fc2, all WMMA.
//
// KFENCE(): empty asm memory fence. Weight fragments are loop-invariant, so
// without it LICM hoists hundreds of global_load_b128 results out of the time
// loops and spills them to scratch (observed in round 1). With the fence the
// fragments are re-fetched from the 192MB L2 right before each WMMA, which is
// the data path this chip rewards.
// ---------------------------------------------------------------------------
#define KFENCE() asm volatile("" ::: "memory")

typedef __attribute__((ext_vector_type(16))) __bf16 bf16x16;
typedef __attribute__((ext_vector_type(8)))  float  f32x8;

union Frag16 { bf16x16 v; unsigned int u[8]; unsigned short h[16]; };

// ----------------------------- problem constants ---------------------------
constexpr int BB  = 256, TIN = 168, TOUT = 24, FF = 64, HH = 128, CC = 8, PER = 8;
constexpr int H4  = 512;               // 4*H (gate width)
constexpr int ENC_FRAGS_PER_C = 416;   // 32 (W0x pad) + 128 (W0h) + 128 (W1x) + 128 (W1h)
constexpr int ENC_FRAGS = CC * ENC_FRAGS_PER_C;      // 3328
constexpr int DEC_FRAGS_PER_F = 288;   // 128 dWh + 32 fc1 + 96 qkv + 32 out
constexpr int DEC_FRAGS = FF * DEC_FRAGS_PER_F;      // 18432

// workspace layout (bytes)
constexpr size_t OFF_ENCW  = 0;
constexpr size_t OFF_ENCB0 = OFF_ENCW  + (size_t)ENC_FRAGS * 512 * 2;
constexpr size_t OFF_ENCB1 = OFF_ENCB0 + (size_t)CC * H4 * 4;
constexpr size_t OFF_DECW  = OFF_ENCB1 + (size_t)CC * H4 * 4;
constexpr size_t OFF_DECB  = OFF_DECW  + (size_t)DEC_FRAGS * 512 * 2;
constexpr size_t OFF_ENCH  = OFF_DECB  + (size_t)FF * H4 * 4;
constexpr size_t OFF_ENCC  = OFF_ENCH  + (size_t)CC * BB * HH * 4;

// LDS strides (halfwords)
constexpr int HS = 136;   // 128 + 8 pad : row-major h/u/k buffers
constexpr int VS = 264;   // 256 + 8 pad : transposed v buffer
constexpr int PS = 40;    // 32 + 8 pad  : per-wave 16x32 A-staging
constexpr size_t SMEM_ENC = (size_t)2 * 16 * HS * 2;
constexpr size_t SMEM_DEC = ((size_t)512 + 3 * 256 * HS + 128 * VS + 16 * 16 * PS) * 2;

// ------------------------------ small helpers ------------------------------
__device__ __forceinline__ unsigned short f2bf(float f) {
  unsigned int x = __float_as_uint(f);
  unsigned int r = x + 0x7fffu + ((x >> 16) & 1u);   // RNE
  return (unsigned short)(r >> 16);
}
__device__ __forceinline__ float bf2f(unsigned short h) {
  return __uint_as_float(((unsigned int)h) << 16);
}
__device__ __forceinline__ float sigm(float x)  { return 1.f / (1.f + __expf(-x)); }
__device__ __forceinline__ float tanh_(float x) { return 2.f * sigm(2.f * x) - 1.f; }
__device__ __forceinline__ float lrelu(float x) { return x > 0.f ? x : 0.01f * x; }

__device__ __forceinline__ f32x8 zero8() {
  f32x8 r;
#pragma unroll
  for (int v = 0; v < 8; ++v) r[v] = 0.f;
  return r;
}

// WMMA wrapper: D = A(16x32 bf16) * B(32x16 bf16) + C(16x16 f32)
__device__ __forceinline__ f32x8 wmma_bf16(bf16x16 a, bf16x16 b, f32x8 c) {
  return __builtin_amdgcn_wmma_f32_16x16x32_bf16(false, a, false, b, (short)0, c,
                                                 false, false);
}

// B fragment from packed global memory (512 bf16 per fragment, lane-major).
__device__ __forceinline__ bf16x16 load_frag_g(const unsigned short* p, int lane) {
  const uint4* q = (const uint4*)p + lane * 2;   // 32 bytes/lane
  uint4 a = q[0], b = q[1];
  Frag16 fr;
  fr.u[0]=a.x; fr.u[1]=a.y; fr.u[2]=a.z; fr.u[3]=a.w;
  fr.u[4]=b.x; fr.u[5]=b.y; fr.u[6]=b.z; fr.u[7]=b.w;
  return fr.v;
}

// A (or B) fragment from row-major bf16 LDS (base = first row of 16-row block).
// ISA 16-bit striping: K = kbase + {0..7 | 16..23} + 8*(lane>=16), row = lane&15.
// -> two contiguous 16B ds_load_b128 per lane.
__device__ __forceinline__ bf16x16 lds_frag(const unsigned short* base, int stride,
                                            int kbase, int lane) {
  int r   = lane & 15;
  int off = kbase + ((lane >> 4) << 3);
  const unsigned short* p = base + r * stride + off;
  uint4 a = *(const uint4*)p;
  uint4 b = *(const uint4*)(p + 16);
  Frag16 fr;
  fr.u[0]=a.x; fr.u[1]=a.y; fr.u[2]=a.z; fr.u[3]=a.w;
  fr.u[4]=b.x; fr.u[5]=b.y; fr.u[6]=b.z; fr.u[7]=b.w;
  return fr.v;
}

// Store a 16x16 f32 D tile (lane -> m = v + 8*(lane>=16), n = lane&15) into
// row-major bf16 LDS at column colbase.
__device__ __forceinline__ void store_tile_lds(unsigned short* base, int stride,
                                               int colbase, f32x8 d, int lane) {
  int col = colbase + (lane & 15);
  int rb  = (lane >> 4) << 3;
#pragma unroll
  for (int v = 0; v < 8; ++v) base[(rb + v) * stride + col] = f2bf(d[v]);
}

// Broadcast per-column bias into a D-layout tile.
__device__ __forceinline__ f32x8 bias_frag(const float* b, int colbase, int lane) {
  float t = b[colbase + (lane & 15)];
  f32x8 r;
#pragma unroll
  for (int v = 0; v < 8; ++v) r[v] = t;
  return r;
}

// Fused LSTM gate combine: c' = sig(f)*c + sig(i)*tanh(g); h = sig(o)*tanh(c')
__device__ __forceinline__ f32x8 lstm_combine(f32x8 gi, f32x8 gf, f32x8 gg,
                                              f32x8 go, f32x8& cst) {
  f32x8 h;
#pragma unroll
  for (int v = 0; v < 8; ++v) {
    float iv = sigm(gi[v]), fv = sigm(gf[v]);
    float gv = tanh_(gg[v]), ov = sigm(go[v]);
    float cn = fv * cst[v] + iv * gv;
    cst[v] = cn;
    h[v] = ov * tanh_(cn);
  }
  return h;
}

// --------------------------- 1) weight packing -----------------------------
__global__ __launch_bounds__(256) void pack_weights_kernel(
    const float* __restrict__ eWih0, const float* __restrict__ eWhh0,
    const float* __restrict__ eWih1, const float* __restrict__ eWhh1,
    const float* __restrict__ dWhh,  const float* __restrict__ fc1w,
    const float* __restrict__ ainw,  const float* __restrict__ aoutw,
    unsigned short* __restrict__ encW, unsigned short* __restrict__ decW) {
  long gid = (long)blockIdx.x * 256 + threadIdx.x;
  const long TOT = (long)(ENC_FRAGS + DEC_FRAGS) * 512;
  if (gid >= TOT) return;
  int frag = (int)(gid >> 9);
  int e    = (int)(gid & 511);
  int lane = e >> 4, j = e & 15;
  int nn = lane & 15;
  int kk = (j < 8 ? j : j + 8) + ((lane >> 4) << 3);   // ISA 16-bit K striping
  float val;
  unsigned short* dst;
  if (frag < ENC_FRAGS) {
    int c = frag / ENC_FRAGS_PER_C, r = frag % ENC_FRAGS_PER_C;
    dst = encW + (size_t)frag * 512 + e;
    if (r < 32) {                    // W0x : (PER=8 -> K padded to 32)
      int n = r * 16 + nn;
      val = (kk < 8) ? eWih0[((size_t)c * H4 + n) * PER + kk] : 0.f;
    } else if (r < 160) {            // W0h^T
      int r2 = r - 32; int nt = r2 >> 2, kt = r2 & 3;
      val = eWhh0[((size_t)c * H4 + nt * 16 + nn) * HH + kt * 32 + kk];
    } else if (r < 288) {            // W1x^T
      int r2 = r - 160; int nt = r2 >> 2, kt = r2 & 3;
      val = eWih1[((size_t)c * H4 + nt * 16 + nn) * HH + kt * 32 + kk];
    } else {                         // W1h^T
      int r2 = r - 288; int nt = r2 >> 2, kt = r2 & 3;
      val = eWhh1[((size_t)c * H4 + nt * 16 + nn) * HH + kt * 32 + kk];
    }
  } else {
    int q = frag - ENC_FRAGS;
    int f = q / DEC_FRAGS_PER_F, r = q % DEC_FRAGS_PER_F;
    dst = decW + (size_t)q * 512 + e;
    if (r < 128) {                   // dec Whh^T
      int nt = r >> 2, kt = r & 3;
      val = dWhh[((size_t)f * H4 + nt * 16 + nn) * HH + kt * 32 + kk];
    } else if (r < 160) {            // fc1^T
      int r2 = r - 128; int nt = r2 >> 2, kt = r2 & 3;
      val = fc1w[((size_t)f * HH + nt * 16 + nn) * HH + kt * 32 + kk];
    } else if (r < 256) {            // Wq / Wk / Wv ^T
      int r2 = r - 160; int w = r2 >> 5, rr = r2 & 31;
      int nt = rr >> 2, kt = rr & 3;
      val = ainw[((size_t)f * 3 * HH + w * HH + nt * 16 + nn) * HH + kt * 32 + kk];
    } else {                         // attn_out^T
      int r2 = r - 256; int nt = r2 >> 2, kt = r2 & 3;
      val = aoutw[((size_t)f * HH + nt * 16 + nn) * HH + kt * 32 + kk];
    }
  }
  *dst = f2bf(val);
}

__global__ __launch_bounds__(256) void pack_bias_kernel(
    const float* __restrict__ b0i, const float* __restrict__ b0h,
    const float* __restrict__ b1i, const float* __restrict__ b1h,
    const float* __restrict__ dbi, const float* __restrict__ dbh,
    float* __restrict__ encB0, float* __restrict__ encB1,
    float* __restrict__ decB) {
  int i = blockIdx.x * 256 + threadIdx.x;
  if (i < CC * H4) { encB0[i] = b0i[i] + b0h[i]; encB1[i] = b1i[i] + b1h[i]; }
  if (i < FF * H4) { decB[i] = dbi[i] + dbh[i]; }
}

// ------------------------------ 2) encoder ---------------------------------
// grid = C * (B/16) single-wave blocks. Each wave: 16 batch rows of one
// cluster through 168 steps of the 2-layer LSTM.
__global__ __launch_bounds__(32) void encoder_kernel(
    const float* __restrict__ x, const unsigned short* __restrict__ encW,
    const float* __restrict__ encB0, const float* __restrict__ encB1,
    float* __restrict__ encH, float* __restrict__ encC) {
  extern __shared__ unsigned short smem[];
  unsigned short* h0b = smem;               // 16 x HS bf16
  unsigned short* h1b = smem + 16 * HS;
  const int lane = threadIdx.x;
  const int c  = blockIdx.x >> 4;
  const int rb = (blockIdx.x & 15) << 4;    // batch row base

  const unsigned short* W0x = encW + (size_t)c * ENC_FRAGS_PER_C * 512;
  const unsigned short* W0h = W0x + 32 * 512;
  const unsigned short* W1x = W0x + 160 * 512;
  const unsigned short* W1h = W0x + 288 * 512;
  const float* b0 = encB0 + c * H4;
  const float* b1 = encB1 + c * H4;

  for (int i = lane; i < 2 * 16 * HS; i += 32) smem[i] = 0;
  __syncthreads();

  f32x8 c0[8], c1[8];
#pragma unroll
  for (int tg = 0; tg < 8; ++tg) { c0[tg] = zero8(); c1[tg] = zero8(); }

  for (int t = 0; t < TIN; ++t) {
    KFENCE();                                  // defeat LICM on weight frags
    __builtin_prefetch((const void*)(W0h), 0, 0);
    // --- x A-fragment (K = 0..7 real features, rest zero-padded) ---
    Frag16 xa;
#pragma unroll
    for (int u = 0; u < 8; ++u) xa.u[u] = 0;
    if (lane < 16) {
      const float* xp = x + ((size_t)(rb + lane) * TIN + t) * FF + c * PER;
#pragma unroll
      for (int j = 0; j < 8; ++j) xa.h[j] = f2bf(xp[j]);
    }
    // --- layer 0 ---
    bf16x16 a0[4];
#pragma unroll
    for (int kt = 0; kt < 4; ++kt) a0[kt] = lds_frag(h0b, HS, kt * 32, lane);
#pragma unroll
    for (int tg = 0; tg < 8; ++tg) {
      KFENCE();                                // bound live B-frag ranges
      f32x8 acc[4];
#pragma unroll
      for (int g = 0; g < 4; ++g) {
        int nt = g * 8 + tg;
        acc[g] = bias_frag(b0, nt * 16, lane);
        acc[g] = wmma_bf16(xa.v, load_frag_g(W0x + (size_t)nt * 512, lane), acc[g]);
#pragma unroll
        for (int kt = 0; kt < 4; ++kt)
          acc[g] = wmma_bf16(a0[kt],
                             load_frag_g(W0h + (size_t)(nt * 4 + kt) * 512, lane),
                             acc[g]);
      }
      f32x8 hh = lstm_combine(acc[0], acc[1], acc[2], acc[3], c0[tg]);
      store_tile_lds(h0b, HS, tg * 16, hh, lane);
    }
    // --- layer 1 ---
    bf16x16 a0n[4], a1[4];
#pragma unroll
    for (int kt = 0; kt < 4; ++kt) {
      a0n[kt] = lds_frag(h0b, HS, kt * 32, lane);
      a1[kt]  = lds_frag(h1b, HS, kt * 32, lane);
    }
#pragma unroll
    for (int tg = 0; tg < 8; ++tg) {
      KFENCE();
      f32x8 acc[4];
#pragma unroll
      for (int g = 0; g < 4; ++g) {
        int nt = g * 8 + tg;
        acc[g] = bias_frag(b1, nt * 16, lane);
#pragma unroll
        for (int kt = 0; kt < 4; ++kt) {
          acc[g] = wmma_bf16(a0n[kt],
                             load_frag_g(W1x + (size_t)(nt * 4 + kt) * 512, lane),
                             acc[g]);
          acc[g] = wmma_bf16(a1[kt],
                             load_frag_g(W1h + (size_t)(nt * 4 + kt) * 512, lane),
                             acc[g]);
        }
      }
      f32x8 hh = lstm_combine(acc[0], acc[1], acc[2], acc[3], c1[tg]);
      store_tile_lds(h1b, HS, tg * 16, hh, lane);
    }
  }
  // --- emit final layer-1 states (f32) ---
  for (int e = lane; e < 16 * HH; e += 32) {
    int r = e >> 7, col = e & 127;
    encH[((size_t)c * BB + rb + r) * HH + col] = bf2f(h1b[r * HS + col]);
  }
  int rb2 = (lane >> 4) << 3;
#pragma unroll
  for (int tg = 0; tg < 8; ++tg)
#pragma unroll
    for (int v = 0; v < 8; ++v)
      encC[((size_t)c * BB + rb + rb2 + v) * HH + tg * 16 + (lane & 15)] = c1[tg][v];
}

// ------------------------------ 3) decoder ---------------------------------
// grid = F blocks x 512 threads (16 waves; wave w owns batch rows 16w..16w+15).
__global__ __launch_bounds__(512) void decoder_kernel(
    const float* __restrict__ x, const unsigned short* __restrict__ decW,
    const float* __restrict__ decB, const float* __restrict__ dWih,
    const float* __restrict__ ainb, const float* __restrict__ aoutb,
    const float* __restrict__ fc1b, const float* __restrict__ fc2w,
    const float* __restrict__ fc2b, const float* __restrict__ encH,
    const float* __restrict__ encC, float* __restrict__ out) {
  extern __shared__ unsigned short smem[];
  float*          xbuf  = (float*)smem;                // 256 f32
  unsigned short* hbuf  = smem + 512;                  // 256 x HS  (h, row-major)
  unsigned short* ubuf  = hbuf + 256 * HS;             // fc1 out, then q
  unsigned short* kbuf  = ubuf + 256 * HS;             // k, row-major
  unsigned short* vbufT = kbuf + 256 * HS;             // v^T : 128 x VS
  unsigned short* pball = vbufT + 128 * VS;            // 16 waves x (16 x PS)

  const int tid = threadIdx.x, lane = tid & 31, wid = tid >> 5;
  const int rowbase = wid * 16;
  const int rb2 = (lane >> 4) << 3;
  unsigned short* pw = pball + wid * 16 * PS;

  const int f = blockIdx.x, cl = f >> 3;
  const unsigned short* Wh  = decW + (size_t)f * DEC_FRAGS_PER_F * 512;
  const unsigned short* Wf1 = Wh + 128 * 512;
  const unsigned short* Wq  = Wh + 160 * 512;
  const unsigned short* Wk  = Wh + 192 * 512;
  const unsigned short* Wv  = Wh + 224 * 512;
  const unsigned short* Wo  = Wh + 256 * 512;
  const float* db = decB + f * H4;
  const float* wx = dWih + f * H4;
  const float* bq = ainb + (size_t)f * 3 * HH;
  const float* bk = bq + HH;
  const float* bv = bq + 2 * HH;
  const float* bo = aoutb + f * HH;
  const float* bf1 = fc1b + f * HH;
  const float* w2 = fc2w + f * HH;
  const float b2 = fc2b[f];

  // --- init: h0 = 0.5*(enc_h[cl] + mean_c(enc_h)); same for c0 ---
  for (int e = tid; e < BB * HH; e += 512) {
    int r = e >> 7, col = e & 127;
    float m = 0.f;
#pragma unroll
    for (int cc = 0; cc < CC; ++cc) m += encH[((size_t)cc * BB + r) * HH + col];
    m *= 0.125f;
    hbuf[r * HS + col] = f2bf(0.5f * (encH[((size_t)cl * BB + r) * HH + col] + m));
  }
  f32x8 cst[8];
#pragma unroll
  for (int tg = 0; tg < 8; ++tg)
#pragma unroll
    for (int v = 0; v < 8; ++v) {
      int row = rowbase + rb2 + v, col = tg * 16 + (lane & 15);
      float m = 0.f;
#pragma unroll
      for (int cc = 0; cc < CC; ++cc) m += encC[((size_t)cc * BB + row) * HH + col];
      m *= 0.125f;
      cst[tg][v] = 0.5f * (encC[((size_t)cl * BB + row) * HH + col] + m);
    }
  if (tid < BB) xbuf[tid] = x[((size_t)tid * TIN + (TIN - 1)) * FF + f];
  __syncthreads();

  for (int ts = 0; ts < TOUT; ++ts) {
    KFENCE();                                  // defeat LICM on weight frags
    __builtin_prefetch((const void*)(Wh), 0, 0);
    // ---- A: 1-step LSTM (own 16 rows) ----
    bf16x16 ah[4];
#pragma unroll
    for (int kt = 0; kt < 4; ++kt)
      ah[kt] = lds_frag(hbuf + rowbase * HS, HS, kt * 32, lane);
    float xr[8];
#pragma unroll
    for (int v = 0; v < 8; ++v) xr[v] = xbuf[rowbase + rb2 + v];
#pragma unroll
    for (int tg = 0; tg < 8; ++tg) {
      KFENCE();
      f32x8 acc[4];
#pragma unroll
      for (int g = 0; g < 4; ++g) {
        int nt = g * 8 + tg, col = nt * 16 + (lane & 15);
        float bb = db[col], ww = wx[col];
#pragma unroll
        for (int v = 0; v < 8; ++v) acc[g][v] = bb + ww * xr[v];   // x @ Wih
#pragma unroll
        for (int kt = 0; kt < 4; ++kt)
          acc[g] = wmma_bf16(ah[kt],
                             load_frag_g(Wh + (size_t)(nt * 4 + kt) * 512, lane),
                             acc[g]);
      }
      f32x8 hh = lstm_combine(acc[0], acc[1], acc[2], acc[3], cst[tg]);
      store_tile_lds(hbuf + rowbase * HS, HS, tg * 16, hh, lane);
    }
    // ---- B: u = lrelu(h @ fc1^T + b1) -> ubuf ----
    bf16x16 ah2[4];
#pragma unroll
    for (int kt = 0; kt < 4; ++kt)
      ah2[kt] = lds_frag(hbuf + rowbase * HS, HS, kt * 32, lane);
#pragma unroll
    for (int nt = 0; nt < 8; ++nt) {
      KFENCE();
      f32x8 acc = bias_frag(bf1, nt * 16, lane);
#pragma unroll
      for (int kt = 0; kt < 4; ++kt)
        acc = wmma_bf16(ah2[kt],
                        load_frag_g(Wf1 + (size_t)(nt * 4 + kt) * 512, lane), acc);
#pragma unroll
      for (int v = 0; v < 8; ++v) acc[v] = lrelu(acc[v]);
      store_tile_lds(ubuf + rowbase * HS, HS, nt * 16, acc, lane);
    }
    // ---- C: k -> kbuf, v -> vbufT (transposed), q -> ubuf (overwrites u) ----
    bf16x16 au[4];
#pragma unroll
    for (int kt = 0; kt < 4; ++kt)
      au[kt] = lds_frag(ubuf + rowbase * HS, HS, kt * 32, lane);
#pragma unroll
    for (int nt = 0; nt < 8; ++nt) {
      KFENCE();
      f32x8 acc = bias_frag(bk, nt * 16, lane);
#pragma unroll
      for (int kt = 0; kt < 4; ++kt)
        acc = wmma_bf16(au[kt],
                        load_frag_g(Wk + (size_t)(nt * 4 + kt) * 512, lane), acc);
      store_tile_lds(kbuf + rowbase * HS, HS, nt * 16, acc, lane);
    }
#pragma unroll
    for (int nt = 0; nt < 8; ++nt) {
      KFENCE();
      f32x8 acc = bias_frag(bv, nt * 16, lane);
#pragma unroll
      for (int kt = 0; kt < 4; ++kt)
        acc = wmma_bf16(au[kt],
                        load_frag_g(Wv + (size_t)(nt * 4 + kt) * 512, lane), acc);
#pragma unroll
      for (int v = 0; v < 8; ++v)      // transposed: vbufT[d][j]
        vbufT[(nt * 16 + (lane & 15)) * VS + rowbase + rb2 + v] = f2bf(acc[v]);
    }
#pragma unroll
    for (int nt = 0; nt < 8; ++nt) {
      KFENCE();
      f32x8 acc = bias_frag(bq, nt * 16, lane);
#pragma unroll
      for (int kt = 0; kt < 4; ++kt)
        acc = wmma_bf16(au[kt],
                        load_frag_g(Wq + (size_t)(nt * 4 + kt) * 512, lane), acc);
      store_tile_lds(ubuf + rowbase * HS, HS, nt * 16, acc, lane);
    }
    __syncthreads();   // publish k, vT (and q)

    // ---- D: S = (q @ k^T) / sqrt(H)  (16 own rows x 256 cols) ----
    bf16x16 aq[4];
#pragma unroll
    for (int kt = 0; kt < 4; ++kt)
      aq[kt] = lds_frag(ubuf + rowbase * HS, HS, kt * 32, lane);
    f32x8 s[16];
#pragma unroll
    for (int nt = 0; nt < 16; ++nt) {
      f32x8 acc = zero8();
#pragma unroll
      for (int kt = 0; kt < 4; ++kt)
        acc = wmma_bf16(aq[kt], lds_frag(kbuf + nt * 16 * HS, HS, kt * 32, lane), acc);
#pragma unroll
      for (int v = 0; v < 8; ++v) s[nt][v] = acc[v] * 0.08838834764831845f;
    }
    // ---- E: row-wise softmax (reduce over 16-lane halves, wave32) ----
#pragma unroll
    for (int v = 0; v < 8; ++v) {
      float mx = s[0][v];
#pragma unroll
      for (int nt = 1; nt < 16; ++nt) mx = fmaxf(mx, s[nt][v]);
#pragma unroll
      for (int d = 1; d < 16; d <<= 1) mx = fmaxf(mx, __shfl_xor(mx, d, 32));
      float su = 0.f;
#pragma unroll
      for (int nt = 0; nt < 16; ++nt) {
        float e = __expf(s[nt][v] - mx);
        s[nt][v] = e;
        su += e;
      }
#pragma unroll
      for (int d = 1; d < 16; d <<= 1) su += __shfl_xor(su, d, 32);
      float inv = 1.f / su;
#pragma unroll
      for (int nt = 0; nt < 16; ++nt) s[nt][v] *= inv;
    }
    // ---- F: O = P @ v  (stage P 32 cols at a time through per-wave LDS) ----
    f32x8 o[8];
#pragma unroll
    for (int n2 = 0; n2 < 8; ++n2) o[n2] = zero8();
#pragma unroll
    for (int kt2 = 0; kt2 < 8; ++kt2) {
      store_tile_lds(pw, PS, 0,  s[2 * kt2],     lane);
      store_tile_lds(pw, PS, 16, s[2 * kt2 + 1], lane);
      bf16x16 ap = lds_frag(pw, PS, 0, lane);
#pragma unroll
      for (int n2 = 0; n2 < 8; ++n2)
        o[n2] = wmma_bf16(ap, lds_frag(vbufT + n2 * 16 * VS, VS, kt2 * 32, lane),
                          o[n2]);
    }
    // ---- G: AO = O @ Wout^T + bout ----
    f32x8 ao[8];
#pragma unroll
    for (int nt = 0; nt < 8; ++nt) ao[nt] = bias_frag(bo, nt * 16, lane);
#pragma unroll
    for (int kt3 = 0; kt3 < 4; ++kt3) {
      KFENCE();
      store_tile_lds(pw, PS, 0,  o[2 * kt3],     lane);
      store_tile_lds(pw, PS, 16, o[2 * kt3 + 1], lane);
      bf16x16 ap = lds_frag(pw, PS, 0, lane);
#pragma unroll
      for (int nt = 0; nt < 8; ++nt)
        ao[nt] = wmma_bf16(ap, load_frag_g(Wo + (size_t)(nt * 4 + kt3) * 512, lane),
                           ao[nt]);
    }
    // ---- H: out = lrelu(AO @ fc2^T + b2); feed back as next x ----
#pragma unroll
    for (int v = 0; v < 8; ++v) {
      float p = 0.f;
#pragma unroll
      for (int nt = 0; nt < 8; ++nt) p += ao[nt][v] * w2[nt * 16 + (lane & 15)];
#pragma unroll
      for (int d = 1; d < 16; d <<= 1) p += __shfl_xor(p, d, 32);
      if ((lane & 15) == 0) {
        int row = rowbase + rb2 + v;
        float val = lrelu(p + b2);
        out[((size_t)row * TOUT + ts) * FF + f] = val;
        xbuf[row] = val;
      }
    }
    __syncthreads();   // publish xbuf; fence k/vT reuse for next step
  }
}

// ------------------------------- launcher ----------------------------------
extern "C" void kernel_launch(void* const* d_in, const int* in_sizes, int n_in,
                              void* d_out, int out_size, void* d_ws, size_t ws_size,
                              hipStream_t stream) {
  (void)in_sizes; (void)n_in; (void)out_size; (void)ws_size;
  const float* x       = (const float*)d_in[0];
  const float* eWih0   = (const float*)d_in[1];
  const float* eWhh0   = (const float*)d_in[2];
  const float* ebih0   = (const float*)d_in[3];
  const float* ebhh0   = (const float*)d_in[4];
  const float* eWih1   = (const float*)d_in[5];
  const float* eWhh1   = (const float*)d_in[6];
  const float* ebih1   = (const float*)d_in[7];
  const float* ebhh1   = (const float*)d_in[8];
  const float* dWih    = (const float*)d_in[9];
  const float* dWhh    = (const float*)d_in[10];
  const float* dbih    = (const float*)d_in[11];
  const float* dbhh    = (const float*)d_in[12];
  const float* ainw    = (const float*)d_in[13];
  const float* ainb    = (const float*)d_in[14];
  const float* aoutw   = (const float*)d_in[15];
  const float* aoutb   = (const float*)d_in[16];
  const float* fc1w    = (const float*)d_in[17];
  const float* fc1b    = (const float*)d_in[18];
  const float* fc2w    = (const float*)d_in[19];
  const float* fc2b    = (const float*)d_in[20];
  float* out = (float*)d_out;

  unsigned char* ws = (unsigned char*)d_ws;
  unsigned short* encW = (unsigned short*)(ws + OFF_ENCW);
  float* encB0 = (float*)(ws + OFF_ENCB0);
  float* encB1 = (float*)(ws + OFF_ENCB1);
  unsigned short* decW = (unsigned short*)(ws + OFF_DECW);
  float* decB  = (float*)(ws + OFF_DECB);
  float* encH  = (float*)(ws + OFF_ENCH);
  float* encC  = (float*)(ws + OFF_ENCC);

  // allow the large dynamic-LDS decoder allocation (host-side, capture-safe)
  (void)hipFuncSetAttribute((const void*)decoder_kernel,
                            hipFuncAttributeMaxDynamicSharedMemorySize,
                            (int)SMEM_DEC);

  const long totalPackElems = (long)(ENC_FRAGS + DEC_FRAGS) * 512;
  int packBlocks = (int)((totalPackElems + 255) / 256);
  pack_weights_kernel<<<packBlocks, 256, 0, stream>>>(
      eWih0, eWhh0, eWih1, eWhh1, dWhh, fc1w, ainw, aoutw, encW, decW);
  pack_bias_kernel<<<(FF * H4 + 255) / 256, 256, 0, stream>>>(
      ebih0, ebhh0, ebih1, ebhh1, dbih, dbhh, encB0, encB1, decB);
  encoder_kernel<<<CC * (BB / 16), 32, SMEM_ENC, stream>>>(
      x, encW, encB0, encB1, encH, encC);
  decoder_kernel<<<FF, 512, SMEM_DEC, stream>>>(
      x, decW, decB, dWih, ainb, aoutb, fc1b, fc2w, fc2b, encH, encC, out);
}
